// MultiHeadAttention_10101763080209
// MI455X (gfx1250) — matrix operand order repaired and checked by reading, not compiled, not run on hardware
//
#include <hip/hip_runtime.h>
#include <hip/hip_bf16.h>

// ---------------------------------------------------------------------------
// MI455X (gfx1250) implementation.
// bf16 WMMA (v_wmma_f32_16x16x32_bf16) with fp32 accumulate everywhere.
// ~44 GFLOP vs ~300MB traffic -> memory floor ~13us @ 23.3TB/s; bf16 matrix
// path keeps compute at/below that floor.
// K-panels of 64 (padded LDS stride 72) halve barrier count vs 32-panels;
// global fills are 128-bit vectorized; epilogues are branch-free (templated).
// ---------------------------------------------------------------------------

typedef __attribute__((ext_vector_type(16))) __bf16 v16bf;
typedef __attribute__((ext_vector_type(8)))  float  v8f;

#define KP  64             // K slice per LDS stage (two wmma deep)
#define LDK 72             // padded LDS row stride (elems) for K=64 panels

__device__ __forceinline__ unsigned short f2bf(float x) {
  unsigned int u = __float_as_uint(x);
  u += 0x7FFFu + ((u >> 16) & 1u);          // round-to-nearest-even
  return (unsigned short)(u >> 16);
}

__device__ __forceinline__ unsigned int pack2bf(float a, float b) {
  return (unsigned int)f2bf(a) | ((unsigned int)f2bf(b) << 16);
}

// A-matrix 16x32 bf16 fragment from LDS (row-major, stride ldk elems).
// ISA layout: lanes 0-15 row M=lane, elems = K{0..7,16..23};
//             lanes 16-31 row M=lane-16, elems = K{8..15,24..31}.
__device__ __forceinline__ v16bf frag_a(const unsigned short* base, int ldk) {
  const int lane = threadIdx.x & 31;
  const int row  = lane & 15;
  const int koff = (lane & 16) ? 8 : 0;
  const unsigned short* p = base + row * ldk + koff;
  union { v16bf v; uint4 q[2]; } r;
  r.q[0] = *reinterpret_cast<const uint4*>(p);        // K koff..koff+7
  r.q[1] = *reinterpret_cast<const uint4*>(p + 16);   // K koff+16..koff+23
  return r.v;
}

// B-matrix 32x16 bf16 fragment; LDS holds B^T row-major as (N,K), stride ldk.
// ISA layout: lanes 0-15 col N=lane, elems = K{0..7,16..23}; lanes 16-31 K{8..15,24..31}.
__device__ __forceinline__ v16bf frag_b(const unsigned short* base, int ldk) {
  const int lane = threadIdx.x & 31;
  const int col  = lane & 15;
  const int koff = (lane & 16) ? 8 : 0;
  const unsigned short* p = base + col * ldk + koff;
  union { v16bf v; uint4 q[2]; } r;
  r.q[0] = *reinterpret_cast<const uint4*>(p);        // K koff..koff+7
  r.q[1] = *reinterpret_cast<const uint4*>(p + 16);   // K koff+16..koff+23
  return r.v;
}

__device__ __forceinline__ v8f wmma_bf16(v16bf a, v16bf b, v8f c) {
  return __builtin_amdgcn_wmma_f32_16x16x32_bf16(
      false, a, false, b, (short)0, c, false, false);
}

__global__ void zero_kernel(float* __restrict__ p, int n) {
  int i = blockIdx.x * blockDim.x + threadIdx.x;
  if (i < n) p[i] = 0.0f;
}

// ---------------------------------------------------------------------------
// Grouped projection: for block n:  C(256x512) = X[:,n,:](256x512) @ W[n](512x512)
// MODE 0: store bf16 into Cbf[(b*64+n)*512 + e]   (qp / kp)
// MODE 1: atomicAdd fp32 into Vacc[b*512 + e]     (sum_w vp -> vsum)
// grid (4 col-tiles, 2 row-tiles, 64 blocks n), 256 threads = 8 waves.
// ---------------------------------------------------------------------------
template <int MODE>
__global__ __launch_bounds__(256)
void proj_kernel(const float* __restrict__ X, const float* __restrict__ W,
                 unsigned short* __restrict__ Cbf, float* __restrict__ Vacc)
{
  __shared__ __attribute__((aligned(16))) unsigned short ldsA[128 * LDK];
  __shared__ __attribute__((aligned(16))) unsigned short ldsB[128 * LDK];

  const int n    = blockIdx.z;
  const int rowb = blockIdx.y * 128;
  const int colb = blockIdx.x * 128;
  const int t    = threadIdx.x;
  const int wave = t >> 5;
  const int wm   = wave >> 1;   // 0..3  -> 32-row band
  const int wn   = wave & 1;    // 0..1  -> 64-col band

  const float* Abase = X + (size_t)rowb * (64 * 512) + (size_t)n * 512;
  const float* Wbase = W + (size_t)n * 512 * 512 + colb;

  v8f acc[2][4] = {};

  for (int k0 = 0; k0 < 512; k0 += KP) {
    __syncthreads();
    // A tile: 128 rows (b) x 64 (d); float4 loads, packed bf16x4 LDS stores
    for (int i = 0; i < 8; ++i) {
      int q = t + i * 256;                     // 0..2047 quads
      int r = q >> 4, dq = (q & 15) * 4;       // 16 quads per row
      float4 v = *reinterpret_cast<const float4*>(
          Abase + (size_t)r * (64 * 512) + k0 + dq);
      uint2 pk = { pack2bf(v.x, v.y), pack2bf(v.z, v.w) };
      *reinterpret_cast<uint2*>(&ldsA[r * LDK + dq]) = pk;
    }
    // B tile: W[d,e] transposed into LDS as (e,d); float4 along e, scattered
    for (int i = 0; i < 8; ++i) {
      int q = t + i * 256;                     // 0..2047 quads
      int d = q >> 5, eq = (q & 31) * 4;       // 32 e-quads per d row
      float4 v = *reinterpret_cast<const float4*>(
          Wbase + (size_t)(k0 + d) * 512 + eq);
      ldsB[(eq + 0) * LDK + d] = f2bf(v.x);
      ldsB[(eq + 1) * LDK + d] = f2bf(v.y);
      ldsB[(eq + 2) * LDK + d] = f2bf(v.z);
      ldsB[(eq + 3) * LDK + d] = f2bf(v.w);
    }
    __syncthreads();

    for (int kp = 0; kp < KP; kp += 32) {
      v16bf af0 = frag_a(ldsA + (wm * 32 +  0) * LDK + kp, LDK);
      v16bf af1 = frag_a(ldsA + (wm * 32 + 16) * LDK + kp, LDK);
      v16bf bf[4];
      for (int j = 0; j < 4; ++j)
        bf[j] = frag_b(ldsB + (wn * 64 + j * 16) * LDK + kp, LDK);
      for (int j = 0; j < 4; ++j) {
        acc[0][j] = wmma_bf16(af0, bf[j], acc[0][j]);
        acc[1][j] = wmma_bf16(af1, bf[j], acc[1][j]);
      }
    }
  }

  const int lane = t & 31;
  const int cN   = lane & 15;
  const int rOff = (lane & 16) ? 8 : 0;   // C layout: vgpr g -> row g or g+8
  for (int i = 0; i < 2; ++i)
    for (int j = 0; j < 4; ++j)
      for (int g = 0; g < 8; ++g) {
        int r = rowb + wm * 32 + i * 16 + rOff + g;
        int e = colb + wn * 64 + j * 16 + cN;
        float v = acc[i][j][g];
        if constexpr (MODE == 0)
          Cbf[((size_t)r * 64 + n) * 512 + e] = f2bf(v);
        else
          atomicAdd(&Vacc[(size_t)r * 512 + e], v);
      }
}

// ---------------------------------------------------------------------------
// Attention tail: one block per (h,b). 4 waves build the 64x64 score tile via
// WMMA (K=64), then scalar max/exp/softmax(rows)/top-16/renorm, then rank-1
// broadcast against vsum -> bf16 `out`.
// ---------------------------------------------------------------------------
__global__ __launch_bounds__(128)
void attn_kernel(const unsigned short* __restrict__ qp,
                 const unsigned short* __restrict__ kpr,
                 const float* __restrict__ vsum,
                 unsigned short* __restrict__ outbf)
{
  __shared__ __attribute__((aligned(16))) unsigned short ldsQ[64 * LDK];
  __shared__ __attribute__((aligned(16))) unsigned short ldsK[64 * LDK];
  __shared__ float s_sc[64 * 64];
  __shared__ float s_a[64];
  __shared__ float s_p[64];
  __shared__ float s_w[64];
  __shared__ float s_f[64];
  __shared__ float s_delta;

  const int hb = blockIdx.x;   // hb = h*B + b (head-major, matches reference)
  const int h  = hb >> 8;
  const int b  = hb & 255;
  const int t  = threadIdx.x;

  // 64x64 bf16 tiles, 128-bit vector copies (8 bf16 per load)
  for (int i = 0; i < 4; ++i) {
    int o = t + i * 128;                 // 0..511 octs
    int r = o >> 3, d8 = (o & 7) * 8;
    *reinterpret_cast<uint4*>(&ldsQ[r * LDK + d8]) =
        *reinterpret_cast<const uint4*>(&qp [((size_t)b * 64 + r) * 512 + h * 64 + d8]);
    *reinterpret_cast<uint4*>(&ldsK[r * LDK + d8]) =
        *reinterpret_cast<const uint4*>(&kpr[((size_t)b * 64 + r) * 512 + h * 64 + d8]);
  }
  __syncthreads();

  const int wave = t >> 5;
  const int r0   = wave * 16;
  v8f acc[4] = {};
  for (int kpan = 0; kpan < 64; kpan += 32) {
    v16bf af = frag_a(ldsQ + r0 * LDK + kpan, LDK);
    for (int j = 0; j < 4; ++j) {
      v16bf bf = frag_b(ldsK + (j * 16) * LDK + kpan, LDK);
      acc[j] = wmma_bf16(af, bf, acc[j]);
    }
  }
  const int lane = t & 31;
  const int cN   = lane & 15;
  const int rOff = (lane & 16) ? 8 : 0;
  for (int j = 0; j < 4; ++j)
    for (int g = 0; g < 8; ++g)
      s_sc[(r0 + rOff + g) * 64 + j * 16 + cN] = acc[j][g];
  __syncthreads();

  // a_r = exp(max_w score / TEMP), TEMP = sqrt(64) = 8
  if (t < 64) {
    float m = -3.4e38f;
    for (int w = 0; w < 64; ++w) m = fmaxf(m, s_sc[t * 64 + w]);
    s_a[t] = __expf(m * 0.125f);
  }
  __syncthreads();
  // softmax over r (64 vals; redundant O(64) scans per lane are cheap)
  if (t < 64) {
    float amax = -3.4e38f;
    for (int j = 0; j < 64; ++j) amax = fmaxf(amax, s_a[j]);
    float sum = 0.0f;
    for (int j = 0; j < 64; ++j) sum += __expf(s_a[j] - amax);
    s_p[t] = __expf(s_a[t] - amax) / sum;
  }
  __syncthreads();
  // top-16 threshold: value with rank 15 (ties broken by index)
  if (t < 64) {
    float pv = s_p[t];
    int rank = 0;
    for (int j = 0; j < 64; ++j) {
      float pj = s_p[j];
      rank += (pj > pv) || (pj == pv && j < t);
    }
    if (rank == 15) s_delta = pv + 1e-7f;
  }
  __syncthreads();
  if (t < 64) s_w[t] = fmaxf(s_p[t] - s_delta, 0.0f);
  __syncthreads();
  if (t < 64) {
    float s = 0.0f;
    for (int j = 0; j < 64; ++j) s += s_w[j];
    s_f[t] = s_w[t] / (s + 1e-7f);
  }
  __syncthreads();

  // out[b,r,h*64+dv] = a[r] * vsum[b,h*64+dv]  (bf16 for the fc/gate GEMM)
  for (int i = 0; i < 8; ++i) {
    int q = t + i * 128;                 // 0..1023 quads
    int r = q >> 4, dv = (q & 15) * 4;
    const float* vs = &vsum[(size_t)b * 512 + h * 64 + dv];
    float a = s_f[r];
    uint2 pk = { pack2bf(a * vs[0], a * vs[1]), pack2bf(a * vs[2], a * vs[3]) };
    *reinterpret_cast<uint2*>(&outbf[((size_t)b * 64 + r) * 512 + h * 64 + dv]) = pk;
  }
}

// ---------------------------------------------------------------------------
// Fused fc + gate GEMMs (16384x512x512 each) sharing the A fragment, with
// sigmoid(gate)*tanh(fc) epilogue to fp32 d_out.
// grid (4 col-tiles, 128 row-tiles), 256 threads.
// ---------------------------------------------------------------------------
__global__ __launch_bounds__(256)
void fcgate_kernel(const unsigned short* __restrict__ Xbf,
                   const float* __restrict__ fc_w, const float* __restrict__ fc_b,
                   const float* __restrict__ gate_w, const float* __restrict__ gate_b,
                   float* __restrict__ out)
{
  __shared__ __attribute__((aligned(16))) unsigned short ldsA[128 * LDK];
  __shared__ __attribute__((aligned(16))) unsigned short ldsF[128 * LDK];
  __shared__ __attribute__((aligned(16))) unsigned short ldsG[128 * LDK];

  const int rowb = blockIdx.y * 128;
  const int colb = blockIdx.x * 128;
  const int t    = threadIdx.x;
  const int wave = t >> 5, wm = wave >> 1, wn = wave & 1;

  v8f accF[2][4] = {}, accG[2][4] = {};

  for (int k0 = 0; k0 < 512; k0 += KP) {
    __syncthreads();
    // A already bf16: 128 rows x 64 e, uint4 copies
    for (int i = 0; i < 4; ++i) {
      int o = t + i * 256;               // 0..1023 octs
      int r = o >> 3, d8 = (o & 7) * 8;
      *reinterpret_cast<uint4*>(&ldsA[r * LDK + d8]) =
          *reinterpret_cast<const uint4*>(&Xbf[(size_t)(rowb + r) * 512 + k0 + d8]);
    }
    // fc_w / gate_w are (o,e) row-major == exactly the (N,K) layout we need
    for (int i = 0; i < 8; ++i) {
      int q = t + i * 256;               // 0..2047 quads
      int r = q >> 4, dq = (q & 15) * 4;
      float4 vf = *reinterpret_cast<const float4*>(
          &fc_w  [(size_t)(colb + r) * 512 + k0 + dq]);
      float4 vg = *reinterpret_cast<const float4*>(
          &gate_w[(size_t)(colb + r) * 512 + k0 + dq]);
      uint2 pf = { pack2bf(vf.x, vf.y), pack2bf(vf.z, vf.w) };
      uint2 pg = { pack2bf(vg.x, vg.y), pack2bf(vg.z, vg.w) };
      *reinterpret_cast<uint2*>(&ldsF[r * LDK + dq]) = pf;
      *reinterpret_cast<uint2*>(&ldsG[r * LDK + dq]) = pg;
    }
    __syncthreads();

    for (int kp = 0; kp < KP; kp += 32) {
      v16bf af0 = frag_a(ldsA + (wm * 32 +  0) * LDK + kp, LDK);
      v16bf af1 = frag_a(ldsA + (wm * 32 + 16) * LDK + kp, LDK);
      v16bf bff[4], bfg[4];
      for (int j = 0; j < 4; ++j) {
        bff[j] = frag_b(ldsF + (wn * 64 + j * 16) * LDK + kp, LDK);
        bfg[j] = frag_b(ldsG + (wn * 64 + j * 16) * LDK + kp, LDK);
      }
      for (int j = 0; j < 4; ++j) {
        accF[0][j] = wmma_bf16(af0, bff[j], accF[0][j]);
        accF[1][j] = wmma_bf16(af1, bff[j], accF[1][j]);
        accG[0][j] = wmma_bf16(af0, bfg[j], accG[0][j]);
        accG[1][j] = wmma_bf16(af1, bfg[j], accG[1][j]);
      }
    }
  }

  const int lane = t & 31, cN = lane & 15, rOff = (lane & 16) ? 8 : 0;
  for (int i = 0; i < 2; ++i)
    for (int j = 0; j < 4; ++j)
      for (int g = 0; g < 8; ++g) {
        int r = rowb + wm * 32 + i * 16 + rOff + g;
        int o = colb + wn * 64 + j * 16 + cN;
        float fv = accF[i][j][g] + fc_b[o];
        float gv = accG[i][j][g] + gate_b[o];
        float sg = 1.0f / (1.0f + __expf(-gv));
        out[(size_t)r * 512 + o] = sg * tanhf(fv);
      }
}

// ---------------------------------------------------------------------------
extern "C" void kernel_launch(void* const* d_in, const int* in_sizes, int n_in,
                              void* d_out, int out_size, void* d_ws, size_t ws_size,
                              hipStream_t stream) {
  const float* q      = (const float*)d_in[0];
  const float* k      = (const float*)d_in[1];
  const float* v      = (const float*)d_in[2];
  const float* Wq     = (const float*)d_in[3];
  const float* Wk     = (const float*)d_in[4];
  const float* Wv     = (const float*)d_in[5];
  const float* fc_w   = (const float*)d_in[6];
  const float* fc_b   = (const float*)d_in[7];
  const float* gate_w = (const float*)d_in[8];
  const float* gate_b = (const float*)d_in[9];
  float* out = (float*)d_out;

  // workspace: qp(16MB bf16) | kp(16MB bf16) | out(16MB bf16) | vsum(512KB f32)
  char* ws = (char*)d_ws;
  unsigned short* qp   = (unsigned short*)(ws);
  unsigned short* kp   = (unsigned short*)(ws + (size_t)16777216);
  unsigned short* obuf = (unsigned short*)(ws + (size_t)33554432);
  float*          vsum = (float*)         (ws + (size_t)50331648);

  zero_kernel<<<512, 256, 0, stream>>>(vsum, 256 * 512);

  dim3 pg(4, 2, 64);   // (e-tiles, b-tiles, blocks n)
  proj_kernel<0><<<pg, 256, 0, stream>>>(q, Wq, qp,      nullptr);
  proj_kernel<0><<<pg, 256, 0, stream>>>(k, Wk, kp,      nullptr);
  proj_kernel<1><<<pg, 256, 0, stream>>>(v, Wv, nullptr, vsum);

  attn_kernel<<<2048, 128, 0, stream>>>(qp, kp, vsum, obuf);

  fcgate_kernel<<<dim3(4, 128), 256, 0, stream>>>(obuf, fc_w, fc_b,
                                                  gate_w, gate_b, out);
}
